// EdgeFeatureGAT_11940009083188
// MI455X (gfx1250) — compile-verified, matrix operand order checked by reading, not run on hardware
//
#include <hip/hip_runtime.h>
#include <cstring>

typedef __attribute__((ext_vector_type(16))) _Float16 v16h;
typedef __attribute__((ext_vector_type(8)))  _Float16 v8h;
typedef __attribute__((ext_vector_type(8)))  float    v8f;

#define HD 128   // hidden width (heads*head_dim == 128 in every layer)

// ---------- ordered-uint float encoding for atomic max ----------
__device__ __forceinline__ unsigned int fl2ord_d(float f) {
  unsigned int u = __float_as_uint(f);
  return (u & 0x80000000u) ? ~u : (u | 0x80000000u);
}
__device__ __forceinline__ float ord2fl_d(unsigned int o) {
  unsigned int u = (o & 0x80000000u) ? (o & 0x7fffffffu) : ~o;
  return __uint_as_float(u);
}

// ---------- fill kernels ----------
__global__ void fill_f32(float* p, float v, int n) {
  int i = blockIdx.x * blockDim.x + threadIdx.x;
  if (i < n) p[i] = v;
}
__global__ void fill_u32(unsigned int* p, unsigned int v, int n) {
  int i = blockIdx.x * blockDim.x + threadIdx.x;
  if (i < n) p[i] = v;
}

// ---------- pre-swizzle weight [K,N](f32) into per-lane WMMA B-fragment layout (f16) ----
// Bh[((nt*(K/32)+kt)*32 + lane)*16 + j] = B[kt*32 + (lane>>4)*8 + (j<8 ? j : 16+(j-8))][nt*16 + (lane&15)]
__global__ void prep_B_kernel(const float* __restrict__ B, _Float16* __restrict__ Bh,
                              int K, int N) {
  int idx = blockIdx.x * blockDim.x + threadIdx.x;
  if (idx >= K * N) return;
  int j    = idx & 15;
  int lane = (idx >> 4) & 31;
  int t    = idx >> 9;              // t = nt*(K/32) + kt
  int kt   = t % (K >> 5);
  int nt   = t / (K >> 5);
  int k = kt * 32 + ((lane >> 4) * 8) + ((j < 8) ? j : 16 + (j - 8));
  int n = nt * 16 + (lane & 15);
  Bh[idx] = (_Float16)B[(size_t)k * N + n];
}

// ---------- small-K encoder: out = relu(LN(x @ W + b)) , width 128 ----------
__global__ void encode_kernel(const float* __restrict__ x, const float* __restrict__ W,
                              const float* __restrict__ bias, const float* __restrict__ g,
                              const float* __restrict__ beta, float* __restrict__ out, int F) {
  __shared__ float xr[16];
  __shared__ float red[HD];
  int n = blockIdx.x, t = threadIdx.x;
  if (t < F) xr[t] = x[(size_t)n * F + t];
  __syncthreads();
  float v = bias[t];
  for (int k = 0; k < F; ++k) v += xr[k] * W[k * HD + t];
  red[t] = v; __syncthreads();
  for (int o = HD / 2; o > 0; o >>= 1) { if (t < o) red[t] += red[t + o]; __syncthreads(); }
  float mean = red[0] * (1.0f / HD);
  __syncthreads();
  float d = v - mean;
  red[t] = d * d; __syncthreads();
  for (int o = HD / 2; o > 0; o >>= 1) { if (t < o) red[t] += red[t + o]; __syncthreads(); }
  float var = red[0] * (1.0f / HD);
  float y = d * rsqrtf(var + 1e-5f) * g[t] + beta[t];
  out[(size_t)n * HD + t] = y > 0.f ? y : 0.f;
}

// ---------- helpers for fragment assembly ----------
__device__ __forceinline__ void cvt8(v16h& d, int lo, const float4& a, const float4& b) {
  d[lo + 0] = (_Float16)a.x; d[lo + 1] = (_Float16)a.y;
  d[lo + 2] = (_Float16)a.z; d[lo + 3] = (_Float16)a.w;
  d[lo + 4] = (_Float16)b.x; d[lo + 5] = (_Float16)b.y;
  d[lo + 6] = (_Float16)b.z; d[lo + 7] = (_Float16)b.w;
}
__device__ __forceinline__ float4 add4(float4 a, float4 b) {
  a.x += b.x; a.y += b.y; a.z += b.z; a.w += b.w; return a;
}

// ---------- WMMA GEMM: C[M,N] = A[M,K] @ B[K,N] (+bias), f32 in/out, f16 compute ----------
// MODE 0: dense A
// MODE 1: A row m = concat( h[row[m]], h[col[m]], ef[m] ), K==384
// MODE 2: A row m = concat( h[row[m]]+gef[m], h[col[m]]+gef[m] ), K==256 (ef param = gef)
template <int MODE>
__global__ void gemm_wmma(const float* __restrict__ A, const _Float16* __restrict__ Bh,
                          const float* __restrict__ bias, float* __restrict__ C,
                          int M, int K, int N,
                          const int* __restrict__ row, const int* __restrict__ col,
                          const float* __restrict__ h, const float* __restrict__ ef) {
  int wave = threadIdx.x >> 5;           // 8 waves per block, stacked in M
  int lane = threadIdx.x & 31;
  int m0   = blockIdx.x * 128 + wave * 16;
  int n0   = blockIdx.y * 16;
  int mrow = m0 + (lane & 15);
  int nn   = n0 + (lane & 15);
  int khalf = (lane >> 4) * 8;           // lanes 0-15: K j+0/16.. ; lanes 16-31: +8
  int mA = (mrow < M) ? mrow : (M - 1);  // clamp: unconditional loads, guarded stores
  int ra = 0, ca = 0;
  if (MODE != 0) { ra = row[mA]; ca = col[mA]; }

  const _Float16* bBase = Bh + (size_t)(n0 >> 4) * (size_t)(K >> 5) * 512 + lane * 16;
  __builtin_prefetch(bBase, 0, 3);

  v8f acc = {};
  for (int k0 = 0; k0 < K; k0 += 32) {
    // ---- A fragment: 8 contiguous floats at khalf, 8 more at khalf+16 (all aligned) ----
    const float* ap;
    if (MODE == 0) {
      ap = A + (size_t)mA * K + k0 + khalf;
    } else if (MODE == 1) {
      int seg = k0 >> 7;   // uniform per iteration: 0 -> h[row], 1 -> h[col], 2 -> ef
      const float* base = (seg == 0) ? (h + (size_t)ra * HD)
                        : (seg == 1) ? (h + (size_t)ca * HD)
                                     : (ef + (size_t)mA * HD);
      ap = base + (k0 & 127) + khalf;
    } else {
      int idx = (k0 < 128) ? ra : ca;    // uniform per iteration
      ap = h + (size_t)idx * HD + (k0 & 127) + khalf;
    }
    float4 a00 = *(const float4*)(ap);
    float4 a01 = *(const float4*)(ap + 4);
    float4 a10 = *(const float4*)(ap + 16);
    float4 a11 = *(const float4*)(ap + 20);
    if (MODE == 2) {
      const float* gp = ef + (size_t)mA * HD + (k0 & 127) + khalf;
      a00 = add4(a00, *(const float4*)(gp));
      a01 = add4(a01, *(const float4*)(gp + 4));
      a10 = add4(a10, *(const float4*)(gp + 16));
      a11 = add4(a11, *(const float4*)(gp + 20));
    }
    v16h af;
    cvt8(af, 0, a00, a01);
    cvt8(af, 8, a10, a11);

    // ---- B fragment: pre-swizzled f16, two contiguous b128 loads ----
    const _Float16* bp = bBase + (size_t)(k0 >> 5) * 512;
    v8h b0 = *(const v8h*)(bp);
    v8h b1 = *(const v8h*)(bp + 8);
    v16h bf;
#pragma unroll
    for (int j = 0; j < 8; ++j) { bf[j] = b0[j]; bf[j + 8] = b1[j]; }

    acc = __builtin_amdgcn_wmma_f32_16x16x32_f16(false, af, false, bf,
                                                 (short)0, acc, false, false);
  }
  float bv = bias ? bias[nn] : 0.f;
  int mbase = m0 + 8 * (lane >> 4);
#pragma unroll
  for (int r = 0; r < 8; ++r) {
    int m = mbase + r;
    if (m < M) C[(size_t)m * N + nn] = acc[r] + bv;
  }
}

// ---------- attention score projections: asrc/adst [N, heads] ----------
__global__ void gat_scores_kernel(const float* __restrict__ x, const float* __restrict__ a_src,
                                  const float* __restrict__ a_dst, float* __restrict__ asrc,
                                  float* __restrict__ adst, int Nn, int heads, int hd) {
  int i = blockIdx.x * blockDim.x + threadIdx.x;
  if (i >= Nn * heads) return;
  int n = i / heads, hh = i % heads;
  const float* xp = x + (size_t)n * HD + hh * hd;
  const float* as = a_src + hh * hd;
  const float* ad = a_dst + hh * hd;
  float s1 = 0.f, s2 = 0.f;
  for (int d = 0; d < hd; ++d) { float xv = xp[d]; s1 += xv * as[d]; s2 += xv * ad[d]; }
  asrc[i] = s1; adst[i] = s2;
}

// ---------- pass 1: e = leaky_relu(asrc[row]+adst[col]), segment max ----------
__global__ void att_pass1(const float* __restrict__ asrc, const float* __restrict__ adst,
                          const int* __restrict__ row, const int* __restrict__ col,
                          float* __restrict__ esc, unsigned int* __restrict__ mOrd,
                          int E, int Nn, int heads) {
  int i = blockIdx.x * blockDim.x + threadIdx.x;
  int total = (E + Nn) * heads;
  if (i >= total) return;
  int eid = i / heads, hh = i % heads;
  int r = (eid < E) ? row[eid] : (eid - E);
  int c = (eid < E) ? col[eid] : (eid - E);
  float v = asrc[r * heads + hh] + adst[c * heads + hh];
  v = (v > 0.f) ? v : 0.2f * v;
  esc[i] = v;
  atomicMax(&mOrd[c * heads + hh], fl2ord_d(v));
}

// ---------- pass 2: ex = exp(e - m[col]); segment sum ----------
__global__ void att_pass2(float* __restrict__ esc, const unsigned int* __restrict__ mOrd,
                          float* __restrict__ ssum, const int* __restrict__ row,
                          const int* __restrict__ col, int E, int Nn, int heads) {
  int i = blockIdx.x * blockDim.x + threadIdx.x;
  int total = (E + Nn) * heads;
  if (i >= total) return;
  int eid = i / heads, hh = i % heads;
  int c = (eid < E) ? col[eid] : (eid - E);
  float m = ord2fl_d(mOrd[c * heads + hh]);
  float ex = __expf(esc[i] - m);
  esc[i] = ex;
  atomicAdd(&ssum[c * heads + hh], ex);
}

// ---------- pass 3: agg[col] += alpha * x[row]  (one block per edge, 128ch) ----------
__global__ void att_aggregate(const float* __restrict__ esc, const float* __restrict__ ssum,
                              const float* __restrict__ x, const int* __restrict__ row,
                              const int* __restrict__ col, float* __restrict__ agg,
                              int E, int Nn, int heads, int hd) {
  int eid = blockIdx.x;
  int ch = threadIdx.x;
  int r = (eid < E) ? row[eid] : (eid - E);
  int c = (eid < E) ? col[eid] : (eid - E);
  int hh = ch / hd;
  float alpha = esc[eid * heads + hh] / ssum[c * heads + hh];
  atomicAdd(&agg[(size_t)c * HD + ch], alpha * x[(size_t)r * HD + ch]);
}

// ---------- h = relu(LN(agg + gat_bias) [+ res]) ----------
__global__ void post_gat_kernel(const float* __restrict__ agg, const float* __restrict__ gbias,
                                const float* __restrict__ g, const float* __restrict__ b,
                                const float* __restrict__ res, float* __restrict__ hout) {
  __shared__ float red[HD];
  int n = blockIdx.x, t = threadIdx.x;
  float v = agg[(size_t)n * HD + t] + gbias[t];
  red[t] = v; __syncthreads();
  for (int o = HD / 2; o > 0; o >>= 1) { if (t < o) red[t] += red[t + o]; __syncthreads(); }
  float mean = red[0] * (1.0f / HD);
  __syncthreads();
  float d = v - mean;
  red[t] = d * d; __syncthreads();
  for (int o = HD / 2; o > 0; o >>= 1) { if (t < o) red[t] += red[t + o]; __syncthreads(); }
  float var = red[0] * (1.0f / HD);
  float y = d * rsqrtf(var + 1e-5f) * g[t] + b[t];
  if (res) y += res[(size_t)n * HD + t];
  hout[(size_t)n * HD + t] = y > 0.f ? y : 0.f;
}

// ---------- in-place relu(LN(z)) for width == blockDim.x (power of two) ----------
__global__ void ln_relu_kernel(float* __restrict__ z, const float* __restrict__ g,
                               const float* __restrict__ b, int Wd) {
  __shared__ float red[HD];
  int n = blockIdx.x, t = threadIdx.x;
  float v = z[(size_t)n * Wd + t];
  red[t] = v; __syncthreads();
  for (int o = Wd / 2; o > 0; o >>= 1) { if (t < o) red[t] += red[t + o]; __syncthreads(); }
  float mean = red[0] / (float)Wd;
  __syncthreads();
  float d = v - mean;
  red[t] = d * d; __syncthreads();
  for (int o = Wd / 2; o > 0; o >>= 1) { if (t < o) red[t] += red[t + o]; __syncthreads(); }
  float var = red[0] / (float)Wd;
  float y = d * rsqrtf(var + 1e-5f) * g[t] + b[t];
  z[(size_t)n * Wd + t] = y > 0.f ? y : 0.f;
}

// ---------- gef = sigmoid(gate_pre) * ef   (in place over gate_pre) ----------
__global__ void gate_apply_kernel(float* __restrict__ gpre, const float* __restrict__ ef, int total) {
  int i = blockIdx.x * blockDim.x + threadIdx.x;
  if (i >= total) return;
  float gp = gpre[i];
  float s = 1.0f / (1.0f + __expf(-gp));
  gpre[i] = s * ef[i];
}

// ---------- final projection: out[E,2] = z2[E,64] @ W3[64,2] + b3 ----------
__global__ void final_proj_kernel(const float* __restrict__ z2, const float* __restrict__ W3,
                                  const float* __restrict__ b3, float* __restrict__ out, int E) {
  int e = blockIdx.x * blockDim.x + threadIdx.x;
  if (e >= E) return;
  const float* z = z2 + (size_t)e * 64;
  float s0 = b3[0], s1 = b3[1];
  for (int k = 0; k < 64; ++k) { float zk = z[k]; s0 += zk * W3[k * 2]; s1 += zk * W3[k * 2 + 1]; }
  out[2 * e] = s0;
  out[2 * e + 1] = s1;
}

static unsigned int h_fl2ord(float f) {
  unsigned int u; memcpy(&u, &f, 4);
  return (u & 0x80000000u) ? ~u : (u | 0x80000000u);
}

extern "C" void kernel_launch(void* const* d_in, const int* in_sizes, int n_in,
                              void* d_out, int out_size, void* d_ws, size_t ws_size,
                              hipStream_t stream) {
  // ---- inputs (setup_inputs dict order, params flattened depth-first) ----
  const float* x       = (const float*)d_in[0];
  const int*   eidx    = (const int*)d_in[1];
  const float* eattr   = (const float*)d_in[2];
  const float* node_W  = (const float*)d_in[3];
  const float* node_b  = (const float*)d_in[4];
  const float* node_lg = (const float*)d_in[5];
  const float* node_lb = (const float*)d_in[6];
  const float* edge_W  = (const float*)d_in[7];
  const float* edge_b  = (const float*)d_in[8];
  const float* edge_lg = (const float*)d_in[9];
  const float* edge_lb = (const float*)d_in[10];
  const float* gate_W  = (const float*)d_in[11];
  const float* gate_b  = (const float*)d_in[12];
  const float* gatW[3]  = {(const float*)d_in[13], (const float*)d_in[17], (const float*)d_in[21]};
  const float* gatAs[3] = {(const float*)d_in[14], (const float*)d_in[18], (const float*)d_in[22]};
  const float* gatAd[3] = {(const float*)d_in[15], (const float*)d_in[19], (const float*)d_in[23]};
  const float* gatB[3]  = {(const float*)d_in[16], (const float*)d_in[20], (const float*)d_in[24]};
  const float* lng[3] = {(const float*)d_in[25], (const float*)d_in[27], (const float*)d_in[29]};
  const float* lnb[3] = {(const float*)d_in[26], (const float*)d_in[28], (const float*)d_in[30]};
  const float* resW[2] = {(const float*)d_in[31], (const float*)d_in[33]};
  const float* resB[2] = {(const float*)d_in[32], (const float*)d_in[34]};
  const float* clf_W1  = (const float*)d_in[35];
  const float* clf_b1  = (const float*)d_in[36];
  const float* clf_l1g = (const float*)d_in[37];
  const float* clf_l1b = (const float*)d_in[38];
  const float* clf_W2  = (const float*)d_in[39];
  const float* clf_b2  = (const float*)d_in[40];
  const float* clf_l2g = (const float*)d_in[41];
  const float* clf_l2b = (const float*)d_in[42];
  const float* clf_W3  = (const float*)d_in[43];
  const float* clf_b3  = (const float*)d_in[44];

  const int N = in_sizes[0] / 10;
  const int E = in_sizes[2] / 10;
  const int Esl = E + N;                 // with self loops
  const int* row = eidx;
  const int* col = eidx + E;
  float* out = (float*)d_out;

  // ---- workspace carve ----
  size_t off = 0;
  auto carve = [&](size_t bytes) -> void* {
    void* p = (char*)d_ws + off;
    off += (bytes + 255) & ~(size_t)255;
    return p;
  };
  float* h     = (float*)carve((size_t)N * HD * 4);
  float* xproj = (float*)carve((size_t)N * HD * 4);
  float* agg   = (float*)carve((size_t)N * HD * 4);
  float* resb  = (float*)carve((size_t)N * HD * 4);
  float* asrc  = (float*)carve((size_t)N * 8 * 4);
  float* adst  = (float*)carve((size_t)N * 8 * 4);
  unsigned int* mOrd = (unsigned int*)carve((size_t)N * 8 * 4);
  float* ssum  = (float*)carve((size_t)N * 8 * 4);
  float* esc   = (float*)carve((size_t)Esl * 8 * 4);
  float* ef    = (float*)carve((size_t)E * HD * 4);   // later reused as z1
  float* gpre  = (float*)carve((size_t)E * HD * 4);   // later gef, then z2
  // pre-swizzled f16 weights
  _Float16* gatWh[3];
  for (int i = 0; i < 3; ++i) gatWh[i] = (_Float16*)carve((size_t)128 * 128 * 2);
  _Float16* resWh[2];
  for (int i = 0; i < 2; ++i) resWh[i] = (_Float16*)carve((size_t)128 * 128 * 2);
  _Float16* gateWh = (_Float16*)carve((size_t)384 * 128 * 2);
  _Float16* clfW1h = (_Float16*)carve((size_t)256 * 128 * 2);
  _Float16* clfW2h = (_Float16*)carve((size_t)128 * 64 * 2);
  (void)ws_size; (void)n_in; (void)out_size;

  // ---- pre-swizzle all GEMM weights into WMMA B-fragment layout ----
  {
    int t;
    t = 128 * 128;
    for (int i = 0; i < 3; ++i)
      prep_B_kernel<<<(t + 255) / 256, 256, 0, stream>>>(gatW[i], gatWh[i], 128, 128);
    for (int i = 0; i < 2; ++i)
      prep_B_kernel<<<(t + 255) / 256, 256, 0, stream>>>(resW[i], resWh[i], 128, 128);
    t = 384 * 128;
    prep_B_kernel<<<(t + 255) / 256, 256, 0, stream>>>(gate_W, gateWh, 384, 128);
    t = 256 * 128;
    prep_B_kernel<<<(t + 255) / 256, 256, 0, stream>>>(clf_W1, clfW1h, 256, 128);
    t = 128 * 64;
    prep_B_kernel<<<(t + 255) / 256, 256, 0, stream>>>(clf_W2, clfW2h, 128, 64);
  }

  const unsigned int NEG_ORD = h_fl2ord(-3.0e38f);
  const int heads_cfg[3] = {8, 8, 1};
  const int hd_cfg[3]    = {16, 16, 128};

  // ---- node encoder ----
  encode_kernel<<<N, HD, 0, stream>>>(x, node_W, node_b, node_lg, node_lb, h, 10);

  // ---- 3 GAT layers ----
  for (int i = 0; i < 3; ++i) {
    int heads = heads_cfg[i], hd = hd_cfg[i];
    // xproj = h @ W
    dim3 gA((N + 127) / 128, HD / 16);
    gemm_wmma<0><<<gA, 256, 0, stream>>>(h, gatWh[i], nullptr, xproj, N, HD, HD,
                                         nullptr, nullptr, nullptr, nullptr);
    // per-node attention projections
    int nh = N * heads;
    gat_scores_kernel<<<(nh + 255) / 256, 256, 0, stream>>>(xproj, gatAs[i], gatAd[i],
                                                            asrc, adst, N, heads, hd);
    // init segment buffers
    fill_u32<<<(nh + 255) / 256, 256, 0, stream>>>(mOrd, NEG_ORD, nh);
    fill_f32<<<(nh + 255) / 256, 256, 0, stream>>>(ssum, 0.f, nh);
    fill_f32<<<((N * HD) + 255) / 256, 256, 0, stream>>>(agg, 0.f, N * HD);
    // segment softmax over incoming edges (incl. implicit self loops)
    int te = Esl * heads;
    att_pass1<<<(te + 255) / 256, 256, 0, stream>>>(asrc, adst, row, col, esc, mOrd, E, N, heads);
    att_pass2<<<(te + 255) / 256, 256, 0, stream>>>(esc, mOrd, ssum, row, col, E, N, heads);
    att_aggregate<<<Esl, HD, 0, stream>>>(esc, ssum, xproj, row, col, agg, E, N, heads, hd);
    // residual branch from h_prev (h not yet overwritten)
    if (i > 0) {
      gemm_wmma<0><<<gA, 256, 0, stream>>>(h, resWh[i - 1], resB[i - 1], resb, N, HD, HD,
                                           nullptr, nullptr, nullptr, nullptr);
    }
    // h = relu(LN(agg + bias) [+ res])
    post_gat_kernel<<<N, HD, 0, stream>>>(agg, gatB[i], lng[i], lnb[i],
                                          (i > 0) ? resb : nullptr, h);
  }

  // ---- edge feature encoder ----
  encode_kernel<<<E, HD, 0, stream>>>(eattr, edge_W, edge_b, edge_lg, edge_lb, ef, 10);

  // ---- gate = sigmoid(concat[h[row],h[col],ef] @ gate_W + gate_b) (gather-fused GEMM) ----
  dim3 gE((E + 127) / 128, HD / 16);
  gemm_wmma<1><<<gE, 256, 0, stream>>>(nullptr, gateWh, gate_b, gpre, E, 384, HD,
                                       row, col, h, ef);
  // gef = sigmoid(gpre) * ef   (in place)
  gate_apply_kernel<<<((E * HD) + 255) / 256, 256, 0, stream>>>(gpre, ef, E * HD);

  // ---- z1 = relu(LN( [h[row]+gef, h[col]+gef] @ clf_W1 + b1 ))  (gather-fused, -> ef) ----
  gemm_wmma<2><<<gE, 256, 0, stream>>>(nullptr, clfW1h, clf_b1, ef, E, 256, HD,
                                       row, col, h, gpre);
  ln_relu_kernel<<<E, HD, 0, stream>>>(ef, clf_l1g, clf_l1b, HD);

  // ---- z2 = relu(LN( z1 @ clf_W2 + b2 ))  (-> gpre, width 64) ----
  dim3 gZ((E + 127) / 128, 64 / 16);
  gemm_wmma<0><<<gZ, 256, 0, stream>>>(ef, clfW2h, clf_b2, gpre, E, HD, 64,
                                       nullptr, nullptr, nullptr, nullptr);
  ln_relu_kernel<<<E, 64, 0, stream>>>(gpre, clf_l2g, clf_l2b, 64);

  // ---- out = z2 @ clf_W3 + b3 ----
  final_proj_kernel<<<(E + 255) / 256, 256, 0, stream>>>(gpre, clf_W3, clf_b3, out, E);
}
